// Forget_GATE_76553497084240
// MI455X (gfx1250) — matrix operand order repaired and checked by reading, not compiled, not run on hardware
//
#include <hip/hip_runtime.h>
#include <math.h>
#include <stdint.h>

// Problem sizes (match reference)
#define B_ 64
#define S_ 2048
#define D_ 256
#define H_ 256

#define KPAD    8
#define USTRIDE (256 + KPAD)   // padded k-stride (elements): 528B rows, 16B aligned, bank-rotating
#define NLOC    128            // H columns per workgroup
#define ROWS    32             // s-rows per double-strip (2 M-tiles per wave)
#define DSTRIPS 4              // 4 x 32 = 128 s-rows per workgroup
#define NCHUNK  16             // S / 128

typedef __bf16  bf16;
typedef __bf16  v8bf  __attribute__((ext_vector_type(8)));
typedef __bf16  v16bf __attribute__((ext_vector_type(16)));
typedef float   v8f   __attribute__((ext_vector_type(8)));
typedef unsigned int u32x4 __attribute__((ext_vector_type(4)));
typedef int          i32x4 __attribute__((ext_vector_type(4)));
typedef int          i32x8 __attribute__((ext_vector_type(8)));

// ---------------- Kernel 1: rec[b,h] = h0[b,:] @ V1[:,h] + B1[h] ----------------
__global__ void rec_kernel(const float* __restrict__ h0,
                           const float* __restrict__ V1,
                           const float* __restrict__ B1,
                           float* __restrict__ rec) {
  int tid = blockIdx.x * blockDim.x + threadIdx.x;   // 0..16383
  int b = tid >> 8, h = tid & 255;
  float s = B1[h];
  const float* hb = h0 + b * H_;
  #pragma unroll 8
  for (int k = 0; k < H_; ++k) s = fmaf(hb[k], V1[k * H_ + h], s);
  rec[tid] = s;
}

// ---- TDM: DMA a [rows x 256] f32 tile (row-major, stride 256) global -> LDS ----
// D# packed per CDNA5 ISA ch.8: group0 = {count, lds_addr, global_addr, type=2},
// group1 = {data_size=4B, tensor_dim0=256, tensor_dim1=S, tile_dim0=256,
//           tile_dim1=rows, tensor_dim0_stride=256}.  Groups 2/3 zero (2D tensor).
__device__ __forceinline__ void tdm_load_tile(uint32_t lds_off, const float* gptr, int rows) {
  uint64_t ga = (uint64_t)(uintptr_t)gptr;
  u32x4 g0 = {0u, 0u, 0u, 0u};
  g0[0] = 1u;                                               // count=1 (valid user D#)
  g0[1] = lds_off;                                          // lds_addr (bytes)
  g0[2] = (uint32_t)ga;                                     // global_addr[31:0]
  g0[3] = (uint32_t)((ga >> 32) & 0x01FFFFFFu) | (2u << 30);// global_addr[56:32] | type=2
  i32x8 g1 = {0, 0, 0, 0, 0, 0, 0, 0};
  g1[0] = (int)(2u << 16);                                  // workgroup_mask=0, data_size=2 (4B)
  g1[1] = (int)(256u << 16);                                // tensor_dim0[15:0]=256
  g1[2] = (int)((uint32_t)S_ << 16);                        // tensor_dim0[31:16]=0 | tensor_dim1[15:0]=2048
  g1[3] = (int)(256u << 16);                                // tensor_dim1[31:16]=0 | tile_dim0=256
  g1[4] = rows & 0xFFFF;                                    // tile_dim1=rows, tile_dim2=0
  g1[5] = 256;                                              // tensor_dim0_stride[31:0]=256
  i32x4 gz = {0, 0, 0, 0};
#if defined(__clang_major__) && (__clang_major__ >= 23)
  i32x8 gz8 = {0, 0, 0, 0, 0, 0, 0, 0};
  __builtin_amdgcn_tensor_load_to_lds(g0, g1, gz, gz, gz8, 0);
#else
  __builtin_amdgcn_tensor_load_to_lds(g0, g1, gz, gz, 0);
#endif
}

__device__ __forceinline__ v16bf ldfragA(const bf16* p) {   // elems {0..7} and {16..23}
  return __builtin_shufflevector(*(const v8bf*)p, *(const v8bf*)(p + 16),
                                 0,1,2,3,4,5,6,7,8,9,10,11,12,13,14,15);
}
__device__ __forceinline__ v16bf ldfragB(const bf16* p) {   // elems {0..7} and {16..23}
  return __builtin_shufflevector(*(const v8bf*)p, *(const v8bf*)(p + 16),
                                 0,1,2,3,4,5,6,7,8,9,10,11,12,13,14,15);
}

// ---------------- Kernel 2: fused GEMM + sigmoid + product-over-S (per chunk) ----------------
// partials[b][chunk][h] = prod_{s in chunk} sigmoid( inp[b,s,:]@U1[:,h] + rec[b,h] )
__global__ __launch_bounds__(256)
void gate_prod_kernel(const float* __restrict__ inp,
                      const float* __restrict__ U1,
                      const float* __restrict__ rec,
                      float* __restrict__ partials) {
  __shared__ bf16  sU_hi[NLOC * USTRIDE];        // U1 slice transposed [n][k], hi/lo split
  __shared__ bf16  sU_lo[NLOC * USTRIDE];        // 2 * 67.5 KB
  __shared__ bf16  sA_hi[ROWS * USTRIDE];        // A double-strip (32 x 256), hi/lo: 33 KB
  __shared__ bf16  sA_lo[ROWS * USTRIDE];
  __shared__ float sAstage[2][ROWS * 256];       // TDM f32 staging, double-buffered: 64 KB

  const int tid   = threadIdx.x;
  const int lane  = tid & 31;
  const int wave  = tid >> 5;          // 0..7, each wave owns one 16-column tile
  const int chunk = blockIdx.x;        // 0..15 : S chunk of 128 rows
  const int hbase = blockIdx.y * NLOC; // 0 or 128
  const int b     = blockIdx.z;        // 0..63
  const int sbase = chunk * (DSTRIPS * ROWS);

  // ---- stage U1 slice (transposed, bf16 hi/lo) ----
  #pragma unroll 4
  for (int i = 0; i < 128; ++i) {
    int idx = i * 256 + tid;           // 0..32767
    int k = idx >> 7;                  // 0..255
    int n = idx & 127;                 // coalesced along n within half-wave
    float u = U1[k * H_ + hbase + n];
    bf16 uh = (bf16)u;
    sU_hi[n * USTRIDE + k] = uh;
    sU_lo[n * USTRIDE + k] = (bf16)(u - (float)uh);
  }

  const int nl = wave * 16 + (lane & 15);          // lane's output column
  const float recv = rec[b * H_ + hbase + nl];
  const float* inpB = inp + (size_t)b * S_ * D_;

  // Fragment addressing (CDNA5 16-bit WMMA layouts):
  //   A: lane holds row lane%16, K = k0 + (lane/16)*8 + {0..7, 16..23}
  //   B: lane holds col lane%16, K = k0 + (lane/16)*8 + {0..7, 16..23}
  const int arow_off = (lane & 15) * USTRIDE + ((lane >> 4) << 3);
  const int brow_off = nl * USTRIDE + ((lane >> 4) << 3);

  // Kick TDM for double-strip 0 (wave 0 only; TENSORcnt tracks it)
  if (tid < 32)
    tdm_load_tile((uint32_t)(uintptr_t)&sAstage[0][0], inpB + (size_t)sbase * D_, ROWS);

  float prodacc = 1.0f;
  int buf = 0;

  for (int ds = 0; ds < DSTRIPS; ++ds) {
    __builtin_amdgcn_s_wait_tensorcnt(0);   // issuer drains its DMA; no-op for other waves
    __syncthreads();                        // stage[buf] visible; prev compute done; U ready (ds==0)

    // ---- convert staged f32 strip -> bf16 hi/lo operand image ----
    #pragma unroll 8
    for (int i = 0; i < ROWS; ++i) {
      float x = sAstage[buf][i * 256 + tid];
      bf16 xh = (bf16)x;
      sA_hi[i * USTRIDE + tid] = xh;
      sA_lo[i * USTRIDE + tid] = (bf16)(x - (float)xh);
    }
    // overlap: DMA the next double-strip into the other stage buffer
    if (ds + 1 < DSTRIPS && tid < 32)
      tdm_load_tile((uint32_t)(uintptr_t)&sAstage[buf ^ 1][0],
                    inpB + (size_t)(sbase + (ds + 1) * ROWS) * D_, ROWS);
    __syncthreads();                        // sA ready

    // ---- two 16x16 tiles (rows 0-15, 16-31) x K=256, split bf16 ----
    v8f acc0 = {0.f,0.f,0.f,0.f,0.f,0.f,0.f,0.f};
    v8f acc1 = {0.f,0.f,0.f,0.f,0.f,0.f,0.f,0.f};
    #pragma unroll
    for (int ks = 0; ks < 8; ++ks) {
      const int k0 = ks * 32;
      v16bf ahi0 = ldfragA(&sA_hi[arow_off + k0]);
      v16bf alo0 = ldfragA(&sA_lo[arow_off + k0]);
      v16bf ahi1 = ldfragA(&sA_hi[arow_off + 16 * USTRIDE + k0]);
      v16bf alo1 = ldfragA(&sA_lo[arow_off + 16 * USTRIDE + k0]);
      v16bf bhi  = ldfragB(&sU_hi[brow_off + k0]);
      v16bf blo  = ldfragB(&sU_lo[brow_off + k0]);
      acc0 = __builtin_amdgcn_wmma_f32_16x16x32_bf16(false, ahi0, false, bhi, (short)0, acc0, false, false);
      acc0 = __builtin_amdgcn_wmma_f32_16x16x32_bf16(false, alo0, false, bhi, (short)0, acc0, false, false);
      acc0 = __builtin_amdgcn_wmma_f32_16x16x32_bf16(false, ahi0, false, blo, (short)0, acc0, false, false);
      acc1 = __builtin_amdgcn_wmma_f32_16x16x32_bf16(false, ahi1, false, bhi, (short)0, acc1, false, false);
      acc1 = __builtin_amdgcn_wmma_f32_16x16x32_bf16(false, alo1, false, bhi, (short)0, acc1, false, false);
      acc1 = __builtin_amdgcn_wmma_f32_16x16x32_bf16(false, ahi1, false, blo, (short)0, acc1, false, false);
    }

    // ---- epilogue: sigmoid, fold both tiles over their 32 rows ----
    float p = 1.0f;
    #pragma unroll
    for (int v = 0; v < 8; ++v) {
      p *= 1.0f / (1.0f + __expf(-(acc0[v] + recv)));
      p *= 1.0f / (1.0f + __expf(-(acc1[v] + recv)));
    }
    p *= __shfl_xor(p, 16, 32);            // combine M halves (wave32)
    prodacc *= p;

    buf ^= 1;
  }

  if (lane < 16)
    partials[((b * NCHUNK + chunk) * H_) + hbase + nl] = prodacc;
}

// ---------------- Kernel 3: out[b,h] = c0[b,h] * prod_chunks partials ----------------
__global__ void finalize_kernel(const float* __restrict__ c0,
                                const float* __restrict__ partials,
                                float* __restrict__ out) {
  int tid = blockIdx.x * blockDim.x + threadIdx.x;   // 0..16383
  int b = tid >> 8, h = tid & 255;
  float p = c0[tid];
  #pragma unroll
  for (int c = 0; c < NCHUNK; ++c) p *= partials[(b * NCHUNK + c) * H_ + h];
  out[tid] = p;
}

extern "C" void kernel_launch(void* const* d_in, const int* in_sizes, int n_in,
                              void* d_out, int out_size, void* d_ws, size_t ws_size,
                              hipStream_t stream) {
  (void)in_sizes; (void)n_in; (void)out_size; (void)ws_size;
  const float* inp = (const float*)d_in[0];
  const float* h0  = (const float*)d_in[1];
  const float* c0  = (const float*)d_in[2];
  const float* U1  = (const float*)d_in[3];
  const float* V1  = (const float*)d_in[4];
  const float* B1  = (const float*)d_in[5];
  float* out = (float*)d_out;

  float* rec      = (float*)d_ws;            // 64*256 floats = 64 KB
  float* partials = rec + B_ * H_;           // 64*16*256 floats = 1 MB

  rec_kernel<<<dim3(64), dim3(256), 0, stream>>>(h0, V1, B1, rec);
  gate_prod_kernel<<<dim3(NCHUNK, 2, B_), dim3(256), 0, stream>>>(inp, U1, rec, partials);
  finalize_kernel<<<dim3(64), dim3(256), 0, stream>>>(c0, partials, out);
}
